// Transformer_1752346657078
// MI455X (gfx1250) — matrix operand order repaired and checked by reading, not compiled, not run on hardware
//
#include <hip/hip_runtime.h>
#include <math.h>

// Problem dims (compile-time constants from the reference)
#define S_LEN 2048
#define D_DIM 1024
#define H_NUM 16
#define HD_DIM 64
#define I_DIM 2816
#define L_NUM 2
#define EPSV 1e-5f

typedef __attribute__((ext_vector_type(16))) __bf16 v16bf;
typedef __attribute__((ext_vector_type(8)))  float  v8f;
typedef __attribute__((ext_vector_type(8)))  unsigned short ush8;
typedef __attribute__((ext_vector_type(4)))  unsigned int u32x4;
typedef __attribute__((ext_vector_type(8)))  int i32x8;
typedef __attribute__((ext_vector_type(4)))  int i32x4;

#if defined(__AMDGCN__) && __has_builtin(__builtin_amdgcn_tensor_load_to_lds) && __has_builtin(__builtin_amdgcn_s_wait_tensorcnt)
#define HAVE_TDM 1
#else
#define HAVE_TDM 0
#endif

union FragBF {
    v16bf bf;
    ush8  u[2];
};

__device__ __forceinline__ unsigned short f2bf(float f) {
    union { float f; unsigned int u; } c; c.f = f;
    unsigned int u = c.u;
    unsigned int r = u + 0x7FFFu + ((u >> 16) & 1u); // round-to-nearest-even
    return (unsigned short)(r >> 16);
}

#if HAVE_TDM
// LDS byte offset of a __shared__ object: low 32 bits of the flat address
// (flat LDS aperture places the LDS byte offset in addr[31:0], ISA 10.2).
__device__ __forceinline__ unsigned lds_addr32(const void* p) {
    return (unsigned)(unsigned long long)p;
}

// Issue one TDM 2-D tile load: tile_dim0=32 bf16 (64B) x tile_dim1=tileRows,
// row stride K elements, with LDS padding 16B after every 64B row so the
// destination matches a [tileRows][40]-ushort padded LDS tile.
// D# layout per CDNA5 ISA section 8 (group0 128b, group1 256b).
// This toolchain's builtin takes 6 args (clang-23 lane):
//   (u32x4 g0, i32x8 g1, i32x4 g2, i32x4 g3, i32x8 g4, i32 cpol)
__device__ __forceinline__ void tdm_load_tile(const unsigned short* gsrc,
                                              unsigned ldsOff, int K, int tileRows) {
    unsigned long long ga = (unsigned long long)gsrc;
    u32x4 g0;
    g0[0] = 1u;                                              // count=1, user mode
    g0[1] = ldsOff;                                          // lds_addr [63:32]
    g0[2] = (unsigned)(ga & 0xFFFFFFFFull);                  // global_addr[31:0]
    g0[3] = (unsigned)((ga >> 32) & 0x1FFFFFFull)            // global_addr[56:32]
          | (2u << 30);                                      // type=2 ("image")
    i32x8 g1;
    g1[0] = (int)((1u << 16)      // data_size = 1 -> 2 bytes
                | (1u << 20)      // pad_enable
                | (3u << 22)      // pad_interval: 16 DWORDs (=64B) before pad
                | (3u << 25));    // pad_amount: 4 DWORDs (=16B) of pad
    g1[1] = (int)(((unsigned)K & 0xFFFFu) << 16);            // tensor_dim0[15:0]
    g1[2] = (int)(((unsigned)tileRows & 0xFFFFu) << 16);     // tensor_dim1[15:0]
    g1[3] = (int)(32u << 16);                                // tile_dim0 = 32
    g1[4] = tileRows;                                        // tile_dim1
    g1[5] = K;                                               // tensor_dim0_stride[31:0]
    g1[6] = 0;
    g1[7] = 0;
    i32x4 gz4 = {0, 0, 0, 0};
    i32x8 gz8 = {0, 0, 0, 0, 0, 0, 0, 0};
    __builtin_amdgcn_tensor_load_to_lds(g0, g1, gz4, gz4, gz8, 0);
}
#endif

// ---------------------------------------------------------------------------
// f32 -> bf16 bulk convert (weights), 4 elems/thread
// ---------------------------------------------------------------------------
__global__ __launch_bounds__(256) void cvt_bf16_kernel(const float* __restrict__ src,
                                                       unsigned short* __restrict__ dst,
                                                       int n) {
    int i = (blockIdx.x * 256 + threadIdx.x) * 4;
    if (i + 3 < n) {
        float4 v = *(const float4*)(src + i);
        dst[i + 0] = f2bf(v.x);
        dst[i + 1] = f2bf(v.y);
        dst[i + 2] = f2bf(v.z);
        dst[i + 3] = f2bf(v.w);
    } else {
        for (int k = i; k < n; ++k) dst[k] = f2bf(src[k]);
    }
}

__global__ __launch_bounds__(256) void copy_f32_kernel(const float* __restrict__ s,
                                                       float* __restrict__ d, int n) {
    int i = blockIdx.x * 256 + threadIdx.x;
    if (i < n) d[i] = s[i];
}

// ---------------------------------------------------------------------------
// RMSNorm: one block per row of D=1024. mode 0 -> bf16 out, mode 1 -> f32 out
// ---------------------------------------------------------------------------
__global__ __launch_bounds__(256) void rmsnorm_kernel(const float* __restrict__ h,
                                                      const float* __restrict__ w,
                                                      unsigned short* __restrict__ obf,
                                                      float* __restrict__ of,
                                                      int mode) {
    __shared__ float red[256];
    const int row = blockIdx.x;
    const int t = threadIdx.x;
    float vals[4];
    float ss = 0.f;
#pragma unroll
    for (int i = 0; i < 4; ++i) {
        vals[i] = h[row * D_DIM + t * 4 + i];
        ss += vals[i] * vals[i];
    }
    red[t] = ss;
    __syncthreads();
    for (int o = 128; o > 0; o >>= 1) {
        if (t < o) red[t] += red[t + o];
        __syncthreads();
    }
    float rs = rsqrtf(red[0] / (float)D_DIM + EPSV);
#pragma unroll
    for (int i = 0; i < 4; ++i) {
        float v = vals[i] * rs * w[t * 4 + i];
        if (mode == 0) obf[row * D_DIM + t * 4 + i] = f2bf(v);
        else           of[row * D_DIM + t * 4 + i] = v;
    }
}

// ---------------------------------------------------------------------------
// RoPE + split qkv(f32, S x 3072) -> q/k (roped) and v, bf16, S x 1024
// ---------------------------------------------------------------------------
__global__ __launch_bounds__(256) void rope_split_kernel(const float* __restrict__ qkv,
                                                         unsigned short* __restrict__ qb,
                                                         unsigned short* __restrict__ kb,
                                                         unsigned short* __restrict__ vb) {
    const int s = blockIdx.x;
    const int t = threadIdx.x;
    const float* row = qkv + s * (3 * D_DIM);
    for (int p = t; p < 512; p += 256) {   // 512 (head, pair) slots
        int i   = p & 31;                  // pair index within head, HD/2 = 32
        int col = p * 2;                   // h*64 + 2*i
        float inv = __powf(10000.0f, -((float)(2 * i)) / 64.0f);
        float ang = (float)s * inv;
        float c, sn;
        __sincosf(ang, &sn, &c);
        float q0 = row[col], q1 = row[col + 1];
        qb[s * D_DIM + col]     = f2bf(q0 * c - q1 * sn);
        qb[s * D_DIM + col + 1] = f2bf(q1 * c + q0 * sn);
        float k0 = row[D_DIM + col], k1 = row[D_DIM + col + 1];
        kb[s * D_DIM + col]     = f2bf(k0 * c - k1 * sn);
        kb[s * D_DIM + col + 1] = f2bf(k1 * c + k0 * sn);
        vb[s * D_DIM + col]     = f2bf(row[2 * D_DIM + col]);
        vb[s * D_DIM + col + 1] = f2bf(row[2 * D_DIM + col + 1]);
    }
}

// ---------------------------------------------------------------------------
// SiLU(a) * b -> bf16, 4 elems/thread
// ---------------------------------------------------------------------------
__global__ __launch_bounds__(256) void silu_mul_kernel(const float* __restrict__ a,
                                                       const float* __restrict__ b,
                                                       unsigned short* __restrict__ o,
                                                       int n) {
    int i = (blockIdx.x * 256 + threadIdx.x) * 4;
#pragma unroll
    for (int k = 0; k < 4; ++k) {
        int idx = i + k;
        if (idx < n) {
            float x = a[idx];
            float s = x / (1.0f + __expf(-x));
            o[idx] = f2bf(s * b[idx]);
        }
    }
}

// ---------------------------------------------------------------------------
// bf16 GEMM: C[M x N] = A[M x K] @ W[N x K]^T  using v_wmma_f32_16x16x32_bf16
// Block: 256 threads (8 waves), tile 128x256, K-step 32, double-buffered LDS.
// Staging: TDM tensor_load_to_lds (wave 0 issues, TENSORcnt-gated), or
// vector global->LDS fallback if the builtin is unavailable.
// Wave grid 2x4 -> each wave computes 64x64 (4x4 fragments of 16x16):
// 16 WMMAs per 16 ds_load_b128 fragment loads per K-step, i.e. 32 FLOP per
// LDS byte, keeping the bf16 XDL pipes fed within the LDS read bandwidth.
// mode 0: C = acc (f32).  mode 1: Hres[m][n] += acc * ls[n].
// ---------------------------------------------------------------------------
__global__ __launch_bounds__(256) void gemm_bf16_kernel(const unsigned short* __restrict__ A,
                                                        const unsigned short* __restrict__ W,
                                                        float* __restrict__ C,
                                                        float* __restrict__ Hres,
                                                        const float* __restrict__ ls,
                                                        int M, int N, int K, int mode) {
    __shared__ unsigned short As[2][128][40];
    __shared__ unsigned short Ws[2][256][40];
    const int bn = blockIdx.x, bm = blockIdx.y;
    const int tid  = threadIdx.x;
    const int wave = tid >> 5, lane = tid & 31;
    const int half = lane >> 4, l16 = lane & 15;
    const int wm = wave & 1, wn = wave >> 1;   // 2 x 4 wave grid (64x64 each)

    const v8f vzero = {0.f, 0.f, 0.f, 0.f, 0.f, 0.f, 0.f, 0.f};
    v8f acc[4][4];
#pragma unroll
    for (int i = 0; i < 4; ++i)
#pragma unroll
        for (int j = 0; j < 4; ++j) acc[i][j] = vzero;

#if HAVE_TDM
    // ---- TDM double-buffered pipeline ----
    const unsigned short* Abase = A + (size_t)bm * 128 * K;
    const unsigned short* Wbase = W + (size_t)bn * 256 * K;
    if (wave == 0) {
        tdm_load_tile(Abase, lds_addr32(&As[0][0][0]), K, 128);
        tdm_load_tile(Wbase, lds_addr32(&Ws[0][0][0]), K, 256);
    }
    int cur = 0;
    for (int k0 = 0; k0 < K; k0 += 32) {
        if (wave == 0) __builtin_amdgcn_s_wait_tensorcnt(0);
        __syncthreads();   // tile `cur` resident; all waves done with `cur^1`
        if (wave == 0 && (k0 + 32) < K) {
            tdm_load_tile(Abase + (k0 + 32), lds_addr32(&As[cur ^ 1][0][0]), K, 128);
            tdm_load_tile(Wbase + (k0 + 32), lds_addr32(&Ws[cur ^ 1][0][0]), K, 256);
        }
        const unsigned short (*pA)[40] = As[cur];
        const unsigned short (*pW)[40] = Ws[cur];
        FragBF a[4], b[4];
#pragma unroll
        for (int i = 0; i < 4; ++i) {
            int m = 64 * wm + 16 * i + l16;
            a[i].u[0] = *(const ush8*)&pA[m][8 * half];
            a[i].u[1] = *(const ush8*)&pA[m][16 + 8 * half];
        }
#pragma unroll
        for (int j = 0; j < 4; ++j) {
            int n = 64 * wn + 16 * j + l16;
            b[j].u[0] = *(const ush8*)&pW[n][8 * half];
            b[j].u[1] = *(const ush8*)&pW[n][16 + 8 * half];
        }
#pragma unroll
        for (int i = 0; i < 4; ++i)
#pragma unroll
            for (int j = 0; j < 4; ++j)
                acc[i][j] = __builtin_amdgcn_wmma_f32_16x16x32_bf16(
                    false, a[i].bf, false, b[j].bf, (short)0, acc[i][j], false, false);
        cur ^= 1;
    }
#else
    // ---- fallback: vector loads + ds stores, single buffer ----
    for (int k0 = 0; k0 < K; k0 += 32) {
        __syncthreads();
        {   // A tile 128x32: 2 threads per row
            int srow = tid >> 1, koff = (tid & 1) * 16;
            int rowA = bm * 128 + srow;
            *(ush8*)&As[0][srow][koff]     = *(const ush8*)&A[rowA * K + k0 + koff];
            *(ush8*)&As[0][srow][koff + 8] = *(const ush8*)&A[rowA * K + k0 + koff + 8];
        }
        {   // W tile 256x32: 1 thread per row
            int rowW = bn * 256 + tid;
#pragma unroll
            for (int c = 0; c < 4; ++c)
                *(ush8*)&Ws[0][tid][8 * c] = *(const ush8*)&W[rowW * K + k0 + 8 * c];
        }
        __syncthreads();
        FragBF a[4], b[4];
#pragma unroll
        for (int i = 0; i < 4; ++i) {
            int m = 64 * wm + 16 * i + l16;
            a[i].u[0] = *(const ush8*)&As[0][m][8 * half];
            a[i].u[1] = *(const ush8*)&As[0][m][16 + 8 * half];
        }
#pragma unroll
        for (int j = 0; j < 4; ++j) {
            int n = 64 * wn + 16 * j + l16;
            b[j].u[0] = *(const ush8*)&Ws[0][n][8 * half];
            b[j].u[1] = *(const ush8*)&Ws[0][n][16 + 8 * half];
        }
#pragma unroll
        for (int i = 0; i < 4; ++i)
#pragma unroll
            for (int j = 0; j < 4; ++j)
                acc[i][j] = __builtin_amdgcn_wmma_f32_16x16x32_bf16(
                    false, a[i].bf, false, b[j].bf, (short)0, acc[i][j], false, false);
    }
#endif

    // Epilogue. C/D layout: VGPR r, lane -> m = r + 8*half, n = l16 (within 16x16)
#pragma unroll
    for (int i = 0; i < 4; ++i)
#pragma unroll
        for (int j = 0; j < 4; ++j) {
            int mbase = bm * 128 + 64 * wm + 16 * i + 8 * half;
            int n     = bn * 256 + 64 * wn + 16 * j + l16;
#pragma unroll
            for (int r = 0; r < 8; ++r) {
                int m = mbase + r;
                float v = acc[i][j][r];
                if (mode == 0) C[m * N + n] = v;
                else           Hres[m * N + n] += v * ls[n];
            }
        }
}

// ---------------------------------------------------------------------------
// Flash attention, causal, per (q-block of 64 rows, head).
// Block: 128 threads = 4 waves; wave w owns q rows [qBase+16w, +16).
// Key loop over 32-key tiles; K tile and transposed V tile staged in LDS.
// QK^T: 2 WMMAs per 16x16 score tile; PV: 4 WMMAs per 32-key step.
// ---------------------------------------------------------------------------
__global__ __launch_bounds__(128) void attn_kernel(const unsigned short* __restrict__ qb,
                                                   const unsigned short* __restrict__ kb,
                                                   const unsigned short* __restrict__ vb,
                                                   unsigned short* __restrict__ yb) {
    __shared__ unsigned short Qs[64][72];
    __shared__ unsigned short Ks[32][72];
    __shared__ unsigned short Vt[64][40];      // transposed: [hd][key]
    __shared__ unsigned short Ps[4][16][40];   // per-wave P scratch

    const int qBase = blockIdx.x * 64;
    const int head  = blockIdx.y;
    const int hoff  = head * HD_DIM;
    const int tid   = threadIdx.x;
    const int wave  = tid >> 5, lane = tid & 31;
    const int half  = lane >> 4, l16 = lane & 15;

    // Stage Q block 64x64
    {
        int r = tid >> 1, doff = (tid & 1) * 32;
        const unsigned short* src = &qb[(qBase + r) * D_DIM + hoff + doff];
#pragma unroll
        for (int c = 0; c < 4; ++c)
            *(ush8*)&Qs[r][doff + 8 * c] = *(const ush8*)&src[8 * c];
    }
    __syncthreads();

    FragBF aq[2];
    {
        int m = 16 * wave + l16;
        aq[0].u[0] = *(const ush8*)&Qs[m][8 * half];
        aq[0].u[1] = *(const ush8*)&Qs[m][16 + 8 * half];
        aq[1].u[0] = *(const ush8*)&Qs[m][32 + 8 * half];
        aq[1].u[1] = *(const ush8*)&Qs[m][48 + 8 * half];
    }

    const v8f vzero = {0.f, 0.f, 0.f, 0.f, 0.f, 0.f, 0.f, 0.f};
    float mrow[8], lrow[8];
    v8f O[4];
#pragma unroll
    for (int r = 0; r < 8; ++r) { mrow[r] = -3.0e38f; lrow[r] = 0.f; }
#pragma unroll
    for (int jc = 0; jc < 4; ++jc) O[jc] = vzero;
    const float scale = 0.125f; // 1/sqrt(64)

    for (int kt = 0; kt < qBase + 64; kt += 32) {
        __syncthreads();
        // Stage K tile 32x64
        {
            int r = tid >> 2, doff = (tid & 3) * 16;
            const unsigned short* src = &kb[(kt + r) * D_DIM + hoff + doff];
            *(ush8*)&Ks[r][doff]     = *(const ush8*)&src[0];
            *(ush8*)&Ks[r][doff + 8] = *(const ush8*)&src[8];
        }
        // Stage V tile transposed 64x32
        {
            int key = tid & 31, dbase = (tid >> 5) * 16;
            const unsigned short* src = &vb[(kt + key) * D_DIM + hoff + dbase];
#pragma unroll
            for (int d = 0; d < 16; ++d) Vt[dbase + d][key] = src[d];
        }
        __syncthreads();

        // Scores: two 16x16 tiles over keys [kt, kt+16), [kt+16, kt+32)
        v8f sc[2];
#pragma unroll
        for (int t = 0; t < 2; ++t) {
            FragBF b0, b1;
            int n = 16 * t + l16;
            b0.u[0] = *(const ush8*)&Ks[n][8 * half];
            b0.u[1] = *(const ush8*)&Ks[n][16 + 8 * half];
            b1.u[0] = *(const ush8*)&Ks[n][32 + 8 * half];
            b1.u[1] = *(const ush8*)&Ks[n][48 + 8 * half];
            v8f s = vzero;
            s = __builtin_amdgcn_wmma_f32_16x16x32_bf16(false, aq[0].bf, false, b0.bf, (short)0, s, false, false);
            s = __builtin_amdgcn_wmma_f32_16x16x32_bf16(false, aq[1].bf, false, b1.bf, (short)0, s, false, false);
            sc[t] = s;
        }

        // Causal mask + online softmax (per row, 16-lane reductions)
#pragma unroll
        for (int r = 0; r < 8; ++r) {
            int qrow = qBase + 16 * wave + r + 8 * half;
            float v0 = (kt + l16      <= qrow) ? sc[0][r] * scale : -3.0e38f;
            float v1 = (kt + 16 + l16 <= qrow) ? sc[1][r] * scale : -3.0e38f;
            float mx = fmaxf(v0, v1);
            for (int o = 1; o < 16; o <<= 1) mx = fmaxf(mx, __shfl_xor(mx, o, 32));
            float newmax = fmaxf(mrow[r], mx);
            float alpha  = __expf(mrow[r] - newmax);
            float p0 = __expf(v0 - newmax);
            float p1 = __expf(v1 - newmax);
            float rsum = p0 + p1;
            for (int o = 1; o < 16; o <<= 1) rsum += __shfl_xor(rsum, o, 32);
            lrow[r] = lrow[r] * alpha + rsum;
            mrow[r] = newmax;
#pragma unroll
            for (int jc = 0; jc < 4; ++jc) O[jc][r] *= alpha;
            // C-layout -> LDS so we can re-read in A-layout
            Ps[wave][r + 8 * half][l16]      = f2bf(p0);
            Ps[wave][r + 8 * half][16 + l16] = f2bf(p1);
        }

        // P (16x32) x V (32x64) -> accumulate O
        FragBF ap;
        ap.u[0] = *(const ush8*)&Ps[wave][l16][8 * half];
        ap.u[1] = *(const ush8*)&Ps[wave][l16][16 + 8 * half];
#pragma unroll
        for (int jc = 0; jc < 4; ++jc) {
            FragBF bv;
            int n = 16 * jc + l16;             // hd column
            bv.u[0] = *(const ush8*)&Vt[n][8 * half];
            bv.u[1] = *(const ush8*)&Vt[n][16 + 8 * half];
            O[jc] = __builtin_amdgcn_wmma_f32_16x16x32_bf16(false, ap.bf, false, bv.bf, (short)0, O[jc], false, false);
        }
    }

    // Finalize: divide by l, store bf16 y
#pragma unroll
    for (int jc = 0; jc < 4; ++jc)
#pragma unroll
        for (int r = 0; r < 8; ++r) {
            int row = qBase + 16 * wave + r + 8 * half;
            int col = hoff + 16 * jc + l16;
            float inv = (lrow[r] > 0.f) ? 1.0f / lrow[r] : 0.f;
            yb[row * D_DIM + col] = f2bf(O[jc][r] * inv);
        }
}

// ---------------------------------------------------------------------------
// Host orchestration
// ---------------------------------------------------------------------------
extern "C" void kernel_launch(void* const* d_in, const int* in_sizes, int n_in,
                              void* d_out, int out_size, void* d_ws, size_t ws_size,
                              hipStream_t stream) {
    const float* x            = (const float*)d_in[0];
    const float* wqkv         = (const float*)d_in[1];
    const float* wo           = (const float*)d_in[2];
    const float* w1           = (const float*)d_in[3];
    const float* w2           = (const float*)d_in[4];
    const float* w3           = (const float*)d_in[5];
    const float* attn_norm_w  = (const float*)d_in[6];
    const float* ffn_norm_w   = (const float*)d_in[7];
    const float* attn_ls      = (const float*)d_in[8];
    const float* ffn_ls       = (const float*)d_in[9];
    const float* final_norm_w = (const float*)d_in[10];
    float* out = (float*)d_out;

    char* ws = (char*)d_ws;
    size_t off = 0;
    auto alloc = [&](size_t bytes) -> void* {
        void* p = ws + off;
        off += (bytes + 255) & ~(size_t)255;
        return p;
    };

    const int SD  = S_LEN * D_DIM;       // 2M
    const int SQ  = S_LEN * 3 * D_DIM;   // 6M
    const int SI  = S_LEN * I_DIM;       // 5.75M

    float*          h    = (float*)alloc((size_t)SD * 4);
    unsigned short* xn   = (unsigned short*)alloc((size_t)SD * 2);
    float*          qkv  = (float*)alloc((size_t)SQ * 4);
    unsigned short* qbuf = (unsigned short*)alloc((size_t)SD * 2);
    unsigned short* kbuf = (unsigned short*)alloc((size_t)SD * 2);
    unsigned short* vbuf = (unsigned short*)alloc((size_t)SD * 2);
    unsigned short* ybuf = (unsigned short*)alloc((size_t)SD * 2);
    float*          f1   = (float*)alloc((size_t)SI * 4);
    float*          f3   = (float*)alloc((size_t)SI * 4);
    unsigned short* fh   = (unsigned short*)alloc((size_t)SI * 2);
    unsigned short* bwqkv = (unsigned short*)alloc((size_t)3 * D_DIM * D_DIM * 2);
    unsigned short* bwo   = (unsigned short*)alloc((size_t)D_DIM * D_DIM * 2);
    unsigned short* bw1   = (unsigned short*)alloc((size_t)I_DIM * D_DIM * 2);
    unsigned short* bw3   = (unsigned short*)alloc((size_t)I_DIM * D_DIM * 2);
    unsigned short* bw2   = (unsigned short*)alloc((size_t)D_DIM * I_DIM * 2);

    // h = x
    copy_f32_kernel<<<(SD + 255) / 256, 256, 0, stream>>>(x, h, SD);

    const int nQKV = 3 * D_DIM * D_DIM;  // 3.1M
    const int nWO  = D_DIM * D_DIM;      // 1.0M
    const int nW1  = I_DIM * D_DIM;      // 2.9M

    for (int l = 0; l < L_NUM; ++l) {
        // Convert this layer's weights to bf16
        cvt_bf16_kernel<<<(nQKV / 4 + 255) / 256, 256, 0, stream>>>(wqkv + (size_t)l * nQKV, bwqkv, nQKV);
        cvt_bf16_kernel<<<(nWO  / 4 + 255) / 256, 256, 0, stream>>>(wo   + (size_t)l * nWO,  bwo,  nWO);
        cvt_bf16_kernel<<<(nW1  / 4 + 255) / 256, 256, 0, stream>>>(w1   + (size_t)l * nW1,  bw1,  nW1);
        cvt_bf16_kernel<<<(nW1  / 4 + 255) / 256, 256, 0, stream>>>(w3   + (size_t)l * nW1,  bw3,  nW1);
        cvt_bf16_kernel<<<(nW1  / 4 + 255) / 256, 256, 0, stream>>>(w2   + (size_t)l * nW1,  bw2,  nW1);

        // xn = rmsnorm(h, attn_norm_w) -> bf16
        rmsnorm_kernel<<<S_LEN, 256, 0, stream>>>(h, attn_norm_w + l * D_DIM, xn, nullptr, 0);

        // qkv = xn @ wqkv^T  (M=2048, N=3072, K=1024)
        gemm_bf16_kernel<<<dim3(3 * D_DIM / 256, S_LEN / 128), 256, 0, stream>>>(
            xn, bwqkv, qkv, nullptr, nullptr, S_LEN, 3 * D_DIM, D_DIM, 0);

        // RoPE + split to bf16 q/k/v
        rope_split_kernel<<<S_LEN, 256, 0, stream>>>(qkv, qbuf, kbuf, vbuf);

        // Flash attention -> ybuf (bf16)
        attn_kernel<<<dim3(S_LEN / 64, H_NUM), 128, 0, stream>>>(qbuf, kbuf, vbuf, ybuf);

        // h += (y @ wo^T) * attn_ls   (M=2048, N=1024, K=1024)
        gemm_bf16_kernel<<<dim3(D_DIM / 256, S_LEN / 128), 256, 0, stream>>>(
            ybuf, bwo, nullptr, h, attn_ls + l * D_DIM, S_LEN, D_DIM, D_DIM, 1);

        // xn = rmsnorm(h, ffn_norm_w)
        rmsnorm_kernel<<<S_LEN, 256, 0, stream>>>(h, ffn_norm_w + l * D_DIM, xn, nullptr, 0);

        // f1 = xn @ w1^T ; f3 = xn @ w3^T  (N=2816, K=1024)
        gemm_bf16_kernel<<<dim3(I_DIM / 256, S_LEN / 128), 256, 0, stream>>>(
            xn, bw1, f1, nullptr, nullptr, S_LEN, I_DIM, D_DIM, 0);
        gemm_bf16_kernel<<<dim3(I_DIM / 256, S_LEN / 128), 256, 0, stream>>>(
            xn, bw3, f3, nullptr, nullptr, S_LEN, I_DIM, D_DIM, 0);

        // fh = silu(f1) * f3 -> bf16
        silu_mul_kernel<<<(SI / 4 + 255) / 256, 256, 0, stream>>>(f1, f3, fh, SI);

        // h += (fh @ w2^T) * ffn_ls  (N=1024, K=2816)
        gemm_bf16_kernel<<<dim3(D_DIM / 256, S_LEN / 128), 256, 0, stream>>>(
            fh, bw2, nullptr, h, ffn_ls + l * D_DIM, S_LEN, D_DIM, I_DIM, 1);
    }

    // out = rmsnorm(h, final_norm_w) -> f32
    rmsnorm_kernel<<<S_LEN, 256, 0, stream>>>(h, final_norm_w, nullptr, out, 1);
}
